// FlashSVDRoPEAttention_85529978733245
// MI455X (gfx1250) — matrix operand order repaired and checked by reading, not compiled, not run on hardware
//
#include <hip/hip_runtime.h>
#include <hip/hip_bf16.h>

typedef __attribute__((ext_vector_type(16))) _Float16 v16h;
typedef __attribute__((ext_vector_type(8)))  _Float16 v8h;
typedef __attribute__((ext_vector_type(8)))  float    v8f;

#define B_  2
#define M_  2048
#define R_  256
#define H_  16
#define DH_ 64
#define D_  (H_ * DH_)          // 1024
#define NEG_BIG (-1e30f)

static constexpr size_t PER = (size_t)B_ * H_ * M_ * DH_;   // 4,194,304 halves per tensor

__device__ __forceinline__ v16h load16(const _Float16* p) {
    union { v16h v; v8h h[2]; } u;
    u.h[0] = *(const v8h*)(p);
    u.h[1] = *(const v8h*)(p + 8);
    return u.v;
}

__device__ __forceinline__ v16h load16_split(const _Float16* lo, const _Float16* hi) {
    union { v16h v; v8h h[2]; } u;
    u.h[0] = *(const v8h*)lo;
    u.h[1] = *(const v8h*)hi;
    return u.v;
}

__device__ __forceinline__ v8f wmma_f16(v16h a, v16h b, v8f c) {
    return __builtin_amdgcn_wmma_f32_16x16x32_f16(false, a, false, b, (short)0, c, false, false);
}

// -----------------------------------------------------------------------------
// Kernel 1: X = P @ V + bias, split into heads; RoPE for q/k; f16 workspace.
// which = blockIdx.x % 3 (0=q, 1=k, 2=v); h = (blockIdx.x/3) % 16; rg = blockIdx.x/48.
// One wave -> one 16-row x 64-col (one head) tile. V-head slab staged in LDS (transposed f16).
// -----------------------------------------------------------------------------
__global__ __launch_bounds__(256) void proj_rope_kernel(
    const float* __restrict__ Pq, const float* __restrict__ Pk, const float* __restrict__ Pv,
    const float* __restrict__ Vq, const float* __restrict__ Vk, const float* __restrict__ Vv,
    const float* __restrict__ bq, const float* __restrict__ bk, const float* __restrict__ bv,
    const int*   __restrict__ pos_ids,
    _Float16* __restrict__ q_ws, _Float16* __restrict__ k_ws, _Float16* __restrict__ v_ws)
{
    constexpr int LSTR = R_ + 8;                 // 264 halves, keeps 16B alignment, spreads banks
    __shared__ _Float16 Vt[DH_ * LSTR];          // 64 * 264 * 2B = 33 KB

    const int tid   = threadIdx.x;
    const int which = blockIdx.x % 3;
    const int h     = (blockIdx.x / 3) % H_;
    const int rg    = blockIdx.x / (3 * H_);     // 0..31

    const float* P    = (which == 0) ? Pq : (which == 1) ? Pk : Pv;
    const float* V    = (which == 0) ? Vq : (which == 1) ? Vk : Vv;
    const float* bias = (which == 0) ? bq : (which == 1) ? bk : bv;

    // Stage V[:, h*64:(h+1)*64] transposed into LDS as f16: Vt[c][r]
    for (int idx = tid; idx < DH_ * R_; idx += 256) {
        int c = idx & (DH_ - 1);
        int r = idx >> 6;
        Vt[c * LSTR + r] = (_Float16)V[(size_t)r * D_ + h * DH_ + c];
    }
    __syncthreads();

    const int wave = tid >> 5;
    const int lane = tid & 31;
    const int ln   = lane & 15;
    const int hi   = lane >> 4;

    const int rt = rg * 8 + wave;                // 0..255 row tiles over B*M
    const int b  = rt >> 7;                      // 128 tiles per batch
    const int m0 = (rt & 127) * 16;

    v8f acc[4] = {};
    const float* prow = P + ((size_t)(b * M_ + m0 + ln)) * R_;

    for (int kk = 0; kk < R_; kk += 32) {
        // A fragment (16x32 f16): halves 0..7 -> K=hi*8+i, 8..15 -> K=16+hi*8+i
        const float4* p4a = (const float4*)(prow + kk + hi * 8);
        const float4* p4b = (const float4*)(prow + kk + 16 + hi * 8);
        float4 x0 = p4a[0], x1 = p4a[1], y0 = p4b[0], y1 = p4b[1];
        v16h a;
        a[0]=(_Float16)x0.x; a[1]=(_Float16)x0.y; a[2]=(_Float16)x0.z; a[3]=(_Float16)x0.w;
        a[4]=(_Float16)x1.x; a[5]=(_Float16)x1.y; a[6]=(_Float16)x1.z; a[7]=(_Float16)x1.w;
        a[8]=(_Float16)y0.x; a[9]=(_Float16)y0.y; a[10]=(_Float16)y0.z; a[11]=(_Float16)y0.w;
        a[12]=(_Float16)y1.x; a[13]=(_Float16)y1.y; a[14]=(_Float16)y1.z; a[15]=(_Float16)y1.w;

        #pragma unroll
        for (int t = 0; t < 4; ++t) {
            // B fragment (32x16): halves 0..7 -> K=hi*8+i, 8..15 -> K=16+hi*8+i (Vt[col][kk+K])
            v16h bf = load16_split(&Vt[(t * 16 + ln) * LSTR + kk + hi * 8], &Vt[(t * 16 + ln) * LSTR + kk + 16 + hi * 8]);
            acc[t] = wmma_f16(a, bf, acc[t]);
        }
    }

    // Bias
    #pragma unroll
    for (int t = 0; t < 4; ++t) {
        float bv4 = bias[h * DH_ + t * 16 + ln];
        #pragma unroll
        for (int j = 0; j < 8; ++j) acc[t][j] += bv4;
    }

    const size_t bh = (size_t)(b * H_ + h);
    if (which == 2) {
        // v: store transposed (B,H,DH,M) so attention B-fragments are contiguous
        #pragma unroll
        for (int t = 0; t < 4; ++t) {
            int dh = t * 16 + ln;
            _Float16* dst = v_ws + (bh * DH_ + dh) * M_ + m0 + hi * 8;
            v8h tmp;
            #pragma unroll
            for (int j = 0; j < 8; ++j) tmp[j] = (_Float16)acc[t][j];
            *(v8h*)dst = tmp;
        }
    } else {
        // q/k: RoPE in-register. dh = t*16+ln; pair (t, t+2) is (dh, dh+32), same lane+slot.
        const float cln  = 0.28782313662425573f;           // ln(10000)/32
        const float inv0 = __expf(-(float)(ln)       * cln);
        const float inv1 = __expf(-(float)(16 + ln)  * cln);
        const float osc  = (which == 0) ? 0.125f : 1.0f;   // fold 1/sqrt(DH) into q
        const int* pp = pos_ids + b * M_ + m0 + hi * 8;
        _Float16* dst_base = ((which == 0) ? q_ws : k_ws) + (bh * M_ + m0 + hi * 8) * DH_;
        #pragma unroll
        for (int j = 0; j < 8; ++j) {
            float pos = (float)pp[j];
            _Float16* dst = dst_base + j * DH_;
            #pragma unroll
            for (int t = 0; t < 2; ++t) {
                float ang = pos * ((t == 0) ? inv0 : inv1);
                float c, s;
                __sincosf(ang, &s, &c);
                float x0 = acc[t][j], x1 = acc[t + 2][j];
                dst[t * 16 + ln]      = (_Float16)((x0 * c - x1 * s) * osc);
                dst[32 + t * 16 + ln] = (_Float16)((x0 * s + x1 * c) * osc);
            }
        }
    }
}

// -----------------------------------------------------------------------------
// Kernel 2: flash attention, block-cooperative.
// One block = one (b,h) and 8 consecutive 16-query tiles (one per wave).
// k/v 32-key tiles staged in LDS once per block via double-buffered
// global_load_async_to_lds_b128 (ASYNCcnt), overlapping fetch with compute.
// -----------------------------------------------------------------------------
__global__ __launch_bounds__(256) void attn_kernel(
    const _Float16* __restrict__ q_ws, const _Float16* __restrict__ k_ws,
    const _Float16* __restrict__ v_ws, const int* __restrict__ amask,
    float* __restrict__ out)
{
    __shared__ _Float16 ktile[2][32 * DH_];      // [key][dh], 4 KB per buffer
    __shared__ _Float16 vtile[2][DH_ * 32];      // [dh][key], 4 KB per buffer
    __shared__ _Float16 lds_p[8][16 * 32];       // per-wave p tile, 1 KB each

    const int tid  = threadIdx.x;
    const int wave = tid >> 5, lane = tid & 31;
    const int ln   = lane & 15, hi = lane >> 4;
    const int bh   = blockIdx.x >> 4;            // 32 (b,h) pairs, 16 blocks each
    const int mg   = blockIdx.x & 15;
    const int m0   = (mg * 8 + wave) * 16;
    const int b    = bh >> 4;

    // Per-thread source pointers for cooperative async staging (one 16B chunk each).
    // k tile: 32 keys x 64 dh halves = 256 chunks; v tile: 64 dh x 32 keys = 256 chunks.
    const _Float16* gk_base = k_ws + ((size_t)bh * M_ + (tid >> 3)) * DH_ + (tid & 7) * 8;
    const _Float16* gv_base = v_ws + ((size_t)bh * DH_ + (tid >> 2)) * M_ + (tid & 3) * 8;
    const unsigned lds_k0 = (unsigned)(size_t)(void*)(&ktile[0][0]) + (unsigned)tid * 16u;
    const unsigned lds_k1 = (unsigned)(size_t)(void*)(&ktile[1][0]) + (unsigned)tid * 16u;
    const unsigned lds_v0 = (unsigned)(size_t)(void*)(&vtile[0][0]) + (unsigned)tid * 16u;
    const unsigned lds_v1 = (unsigned)(size_t)(void*)(&vtile[1][0]) + (unsigned)tid * 16u;

#define ISSUE_STAGE(bufi, nb2)                                                          \
    do {                                                                                \
        unsigned long long gka = (unsigned long long)(size_t)(gk_base + (size_t)(nb2) * DH_); \
        unsigned long long gva = (unsigned long long)(size_t)(gv_base + (nb2));         \
        unsigned lka = (bufi) ? lds_k1 : lds_k0;                                        \
        unsigned lva = (bufi) ? lds_v1 : lds_v0;                                        \
        asm volatile("global_load_async_to_lds_b128 %0, %1, off"                        \
                     :: "v"(lka), "v"(gka) : "memory");                                 \
        asm volatile("global_load_async_to_lds_b128 %0, %1, off"                        \
                     :: "v"(lva), "v"(gva) : "memory");                                 \
    } while (0)

    // q A-fragments: row = ln, K halves: 16 contiguous at K=hi*16
    const _Float16* qrow = q_ws + ((size_t)bh * M_ + m0 + ln) * DH_;
    v16h aq0 = load16(qrow + hi * 16);
    v16h aq1 = load16(qrow + 32 + hi * 16);

    float rm[8], rl[8];
    #pragma unroll
    for (int j = 0; j < 8; ++j) { rm[j] = NEG_BIG; rl[j] = 0.0f; }
    v8f acc[4] = {};
    _Float16* pl = lds_p[wave];

    // Prologue: start fetching the first key tile.
    ISSUE_STAGE(0, 0);

    int buf = 0;
    for (int nb = 0; nb < M_; nb += 32, buf ^= 1) {
        if (nb + 32 < M_) {
            ISSUE_STAGE(buf ^ 1, nb + 32);                   // prefetch next tile
            asm volatile("s_wait_asynccnt 0x2" ::: "memory"); // current tile's loads done
        } else {
            asm volatile("s_wait_asynccnt 0x0" ::: "memory");
        }
        __syncthreads();                                      // all waves' chunks visible

        const _Float16* kb = ktile[buf];
        const _Float16* vb = vtile[buf];

        // scores: 16x32 keys, K=DH in two 32-chunks
        v16h bk0l = load16(kb + ln * DH_ + hi * 16);
        v16h bk0h = load16(kb + ln * DH_ + 32 + hi * 16);
        v16h bk1l = load16(kb + (16 + ln) * DH_ + hi * 16);
        v16h bk1h = load16(kb + (16 + ln) * DH_ + 32 + hi * 16);

        v8f s0 = {}, s1 = {};
        s0 = wmma_f16(aq0, bk0l, s0);
        s0 = wmma_f16(aq1, bk0h, s0);
        s1 = wmma_f16(aq0, bk1l, s1);
        s1 = wmma_f16(aq1, bk1h, s1);

        const int mk0 = amask[b * M_ + nb + ln];
        const int mk1 = amask[b * M_ + nb + 16 + ln];
        const float pm0 = mk0 ? 1.0f : 0.0f;
        const float pm1 = mk1 ? 1.0f : 0.0f;
        #pragma unroll
        for (int j = 0; j < 8; ++j) {
            if (!mk0) s0[j] = NEG_BIG;
            if (!mk1) s1[j] = NEG_BIG;
        }

        // online softmax (row = hi*8+j lives within one 16-lane half)
        #pragma unroll
        for (int j = 0; j < 8; ++j) {
            float tm = fmaxf(s0[j], s1[j]);
            tm = fmaxf(tm, __shfl_xor(tm, 1, 32));
            tm = fmaxf(tm, __shfl_xor(tm, 2, 32));
            tm = fmaxf(tm, __shfl_xor(tm, 4, 32));
            tm = fmaxf(tm, __shfl_xor(tm, 8, 32));
            float nm   = fmaxf(rm[j], tm);
            float corr = __expf(rm[j] - nm);
            rm[j] = nm;
            float p0 = __expf(s0[j] - nm) * pm0;
            float p1 = __expf(s1[j] - nm) * pm1;
            s0[j] = p0; s1[j] = p1;
            float rs = p0 + p1;
            rs += __shfl_xor(rs, 1, 32);
            rs += __shfl_xor(rs, 2, 32);
            rs += __shfl_xor(rs, 4, 32);
            rs += __shfl_xor(rs, 8, 32);
            rl[j] = rl[j] * corr + rs;
            #pragma unroll
            for (int t = 0; t < 4; ++t) acc[t][j] *= corr;
        }

        // p tile: C-layout -> per-wave LDS -> A-layout
        #pragma unroll
        for (int j = 0; j < 8; ++j) {
            int row = hi * 8 + j;
            pl[row * 32 + ln]      = (_Float16)s0[j];
            pl[row * 32 + 16 + ln] = (_Float16)s1[j];
        }
        asm volatile("s_wait_dscnt 0" ::: "memory");
        v16h ap = load16(pl + ln * 32 + hi * 16);

        // out += p @ v  (v tile transposed in LDS: contiguous over keys)
        #pragma unroll
        for (int t = 0; t < 4; ++t) {
            v16h bv = load16(vb + (t * 16 + ln) * 32 + hi * 16);
            acc[t] = wmma_f16(ap, bv, acc[t]);
        }

        __syncthreads();   // all waves done reading buf before it is refilled (stage nb+64)
    }
#undef ISSUE_STAGE

    // Normalize with l>0 guard (matches reference where(l>0, out/l, 0))
    #pragma unroll
    for (int j = 0; j < 8; ++j) {
        float l   = rl[j];
        float inv = (l > 0.0f) ? 1.0f / l : 0.0f;
        #pragma unroll
        for (int t = 0; t < 4; ++t) acc[t][j] *= inv;
    }
    #pragma unroll
    for (int t = 0; t < 4; ++t) {
        #pragma unroll
        for (int j = 0; j < 8; ++j) {
            out[((size_t)bh * M_ + m0 + hi * 8 + j) * DH_ + t * 16 + ln] = acc[t][j];
        }
    }
}

extern "C" void kernel_launch(void* const* d_in, const int* in_sizes, int n_in,
                              void* d_out, int out_size, void* d_ws, size_t ws_size,
                              hipStream_t stream) {
    const float* Pq = (const float*)d_in[0];
    const float* Pk = (const float*)d_in[1];
    const float* Pv = (const float*)d_in[2];
    const float* Vq = (const float*)d_in[3];
    const float* Vk = (const float*)d_in[4];
    const float* Vv = (const float*)d_in[5];
    const float* bq = (const float*)d_in[6];
    const float* bk = (const float*)d_in[7];
    const float* bv = (const float*)d_in[8];
    const int* amask = (const int*)d_in[9];
    const int* pos   = (const int*)d_in[10];
    float* out = (float*)d_out;

    _Float16* q_ws = (_Float16*)d_ws;
    _Float16* k_ws = q_ws + PER;
    _Float16* v_ws = k_ws + PER;

    // 32 row-groups * 16 heads * 3 tensors = 1536 blocks of 256 threads (8 waves)
    proj_rope_kernel<<<dim3(1536), dim3(256), 0, stream>>>(
        Pq, Pk, Pv, Vq, Vk, Vv, bq, bk, bv, pos, q_ws, k_ws, v_ws);

    // 32 (b,h) * 16 query-tile groups = 512 blocks of 256 threads (8 waves)
    attn_kernel<<<dim3(512), dim3(256), 0, stream>>>(q_ws, k_ws, v_ws, amask, out);
}
